// _PurePyMamba_56109452755532
// MI455X (gfx1250) — compile-verified, matrix operand order checked by reading, not run on hardware
//
#include <hip/hip_runtime.h>

// ---- problem constants (match reference) ----
#define D_MODEL  1024
#define D_STATE  16
#define D_CONV   4
#define D_INNER  2048      // EXPAND * D_MODEL
#define BATCH    2
#define SEQ      1024
#define M_TOT    (BATCH * SEQ)   // 2048 rows of activations
#define N_XZ     (2 * D_INNER)   // 4096
#define N_SSM    33              // dt_raw + B(16) + C(16)
#define N_SSM_PAD 64             // padded for WMMA tiling

// ---- GEMM tiling ----
#define M_BLK   64               // rows per block (4 waves x 16)
#define N_BLK   128              // cols per block (2 wave-cols x 64)
#define K_STEP  64               // two 16x16x32 WMMA K-chunks per LDS slab
#define LDSB    80               // LDS row pitch in bf16: chunk0 @ +0, pad 16B, chunk1 @ +40, pad 16B

typedef __bf16 bf16;
typedef __attribute__((ext_vector_type(16))) __bf16 v16bf;
typedef __attribute__((ext_vector_type(8)))  __bf16 v8bf;
typedef __attribute__((ext_vector_type(8)))  float  v8f;
typedef __attribute__((ext_vector_type(4)))  unsigned int u32x4;
typedef __attribute__((ext_vector_type(4)))  int i32x4;
typedef __attribute__((ext_vector_type(8)))  int i32x8;

#if __has_builtin(__builtin_amdgcn_tensor_load_to_lds)
#define HAVE_TDM 1
#else
#define HAVE_TDM 0
#endif

// round-to-nearest-even f32 -> bf16
__device__ __forceinline__ bf16 f2bf(float f) {
    unsigned u = __builtin_bit_cast(unsigned, f);
    u += 0x7fffu + ((u >> 16) & 1u);
    unsigned short s = (unsigned short)(u >> 16);
    return __builtin_bit_cast(bf16, s);
}

// Assemble a 16x32 bf16 WMMA fragment from a row at p (per-lane pointer).
// lanes 0..15 : K = {0..7} U {16..23}; lanes 16..31 : K = {8..15} U {24..31}
__device__ __forceinline__ v16bf frag_from(const bf16* p) {
    v8bf lo = *(const v8bf*)(p);
    v8bf hi = *(const v8bf*)(p + 16);
    v16bf f;
#pragma unroll
    for (int i = 0; i < 8; ++i) { f[i] = lo[i]; f[i + 8] = hi[i]; }
    return f;
}

// Issue a TDM 2D tile load: tile_rows x 64 bf16 elements of a row-major
// [rows_tensor, K] matrix starting at gsrc, into LDS at lds_off. TDM padding
// (pad_interval=16 DWORDs, pad_amount=4 DWORDs) yields a 160-byte LDS row
// pitch: k-chunk0 at +0B, k-chunk1 at +80B (element offset 40).
__device__ __forceinline__ void tdm_load_tile(const bf16* gsrc, unsigned lds_off,
                                              int rows_tensor, int K, int tile_rows) {
#if HAVE_TDM
    unsigned long long ga = (unsigned long long)(uintptr_t)gsrc;
    // group0: count=1 | lds_addr | global_addr[56:0] | type=2 (bits 127:126)
    u32x4 g0 = { 1u,
                 lds_off,
                 (unsigned)ga,
                 (unsigned)((ga >> 32) & 0x01FFFFFFull) | 0x80000000u };
    // group1 word0: data_size=1(2B)<<16 | pad_enable<<20 | pad_interval=3(16dw)<<22 | pad_amount=3(4dw)<<25
    i32x8 g1 = { (int)0x06D10000u,
                 (int)(((unsigned)K & 0xFFFFu) << 16),                                   // tensor_dim0[15:0]
                 (int)(((unsigned)K >> 16) | (((unsigned)rows_tensor & 0xFFFFu) << 16)), // dim0[31:16] | dim1[15:0]
                 (int)(((unsigned)rows_tensor >> 16) | ((unsigned)K_STEP << 16)),        // dim1[31:16] | tile_dim0=64
                 (int)(unsigned)tile_rows,                                               // tile_dim1 | tile_dim2=0
                 (int)K,                                                                 // tensor_dim0_stride[31:0]
                 0, 0 };
    i32x4 z4 = { 0, 0, 0, 0 };
#if defined(__clang_major__) && (__clang_major__ >= 23)
    i32x8 z8 = { 0, 0, 0, 0, 0, 0, 0, 0 };
    __builtin_amdgcn_tensor_load_to_lds(g0, g1, z4, z4, z8, 0);
#else
    __builtin_amdgcn_tensor_load_to_lds(g0, g1, z4, z4, 0);
#endif
#else
    (void)gsrc; (void)lds_off; (void)rows_tensor; (void)K; (void)tile_rows;
#endif
}

// ---------------- bf16 WMMA GEMM: C[M,N] = A[M,K] * B[N,K]^T ----------------
// block = 256 threads (8 waves = 4 M-waves x 2 N-waves); block tile 64 x 128;
// wave tile 16 x 64 (4 accumulators, 8 WMMAs per K-slab). K-slabs are staged
// in double-buffered LDS by the Tensor Data Mover so the DMA of slab k+1
// overlaps the WMMAs on slab k (s_wait_tensorcnt depth 2).
__global__ __launch_bounds__(256) void gemm_bf16_wmma(
    const bf16* __restrict__ A, const bf16* __restrict__ B, float* __restrict__ C,
    int M, int N, int K)
{
    __shared__ __align__(128) bf16 sA[2][M_BLK * LDSB];   // 2 x 10 KB
    __shared__ __align__(128) bf16 sB[2][N_BLK * LDSB];   // 2 x 20 KB

    const int lane = threadIdx.x & 31;
    const int wave = threadIdx.x >> 5;
    const int wm   = wave & 3;           // 0..3
    const int wn   = wave >> 2;          // 0..1
    const int mBlk = blockIdx.y * M_BLK;
    const int nBlk = blockIdx.x * N_BLK;

    v8f acc[4];
#pragma unroll
    for (int j = 0; j < 4; ++j)
#pragma unroll
        for (int r = 0; r < 8; ++r) acc[j][r] = 0.f;

    const int mloc = wm * 16 + (lane & 15);
    const int kh   = (lane >> 4) << 3;    // 0 or 8
    const int ksteps = K / K_STEP;

#if HAVE_TDM
    if (wave == 0) {   // prologue: slab 0 -> buffer 0
        tdm_load_tile(A + (size_t)mBlk * K, (unsigned)(uintptr_t)&sA[0][0], M, K, M_BLK);
        tdm_load_tile(B + (size_t)nBlk * K, (unsigned)(uintptr_t)&sB[0][0], N, K, N_BLK);
    }
#endif

    for (int ks = 0; ks < ksteps; ++ks) {
        const int cur = ks & 1;
#if HAVE_TDM
        if (wave == 0) {
            if (ks + 1 < ksteps) {
                // issue slab ks+1 into the other buffer (it was last read in
                // step ks-1; the end-of-step barrier already guaranteed that),
                // then wait only for slab ks's two descriptors.
                const int nxt = 1 - cur;
                tdm_load_tile(A + (size_t)mBlk * K + (size_t)(ks + 1) * K_STEP,
                              (unsigned)(uintptr_t)&sA[nxt][0], M, K, M_BLK);
                tdm_load_tile(B + (size_t)nBlk * K + (size_t)(ks + 1) * K_STEP,
                              (unsigned)(uintptr_t)&sB[nxt][0], N, K, N_BLK);
                __builtin_amdgcn_s_wait_tensorcnt(2);
            } else {
                __builtin_amdgcn_s_wait_tensorcnt(0);
            }
        }
#else
        {   // cooperative fill of buffer `cur`, identical padded layout
            const int t = threadIdx.x;
#pragma unroll
            for (int task = t; task < 512; task += 256) {   // B: 128 rows x 4 chunks
                const int row = task >> 2, q = task & 3;
                bf16* dst = &sB[cur][row * LDSB + (q >> 1) * 40 + (q & 1) * 16];
                if (nBlk + row < N) {
                    const bf16* src = B + (size_t)(nBlk + row) * K + ks * K_STEP + q * 16;
                    *(v8bf*)dst       = *(const v8bf*)src;
                    *(v8bf*)(dst + 8) = *(const v8bf*)(src + 8);
                } else {
                    v8bf z;
#pragma unroll
                    for (int i = 0; i < 8; ++i) z[i] = f2bf(0.f);
                    *(v8bf*)dst = z; *(v8bf*)(dst + 8) = z;
                }
            }
            {   // A: 64 rows x 4 chunks = 256 tasks
                const int row = t >> 2, q = t & 3;
                const bf16* src = A + (size_t)(mBlk + row) * K + ks * K_STEP + q * 16;
                bf16* dst = &sA[cur][row * LDSB + (q >> 1) * 40 + (q & 1) * 16];
                *(v8bf*)dst       = *(const v8bf*)src;
                *(v8bf*)(dst + 8) = *(const v8bf*)(src + 8);
            }
        }
#endif
        __syncthreads();   // slab `cur` visible to all waves

        const bf16* bufA = &sA[cur][0];
        const bf16* bufB = &sB[cur][0];
#pragma unroll
        for (int half = 0; half < 2; ++half) {
            const int off = half * 40;   // k-chunk element offset within row
            v16bf a = frag_from(&bufA[mloc * LDSB + off + kh]);
#pragma unroll
            for (int j = 0; j < 4; ++j) {
                v16bf b = frag_from(&bufB[(wn * 64 + j * 16 + (lane & 15)) * LDSB + off + kh]);
                acc[j] = __builtin_amdgcn_wmma_f32_16x16x32_bf16(
                    false, a, false, b, (short)0, acc[j], false, false);
            }
        }
        __syncthreads();   // all waves done reading `cur` -> safe to refill
    }

    // C/D layout: lanes 0..15 -> N=lane, M=r; lanes 16..31 -> N=lane-16, M=8+r
    const int col = lane & 15;
    const int rb  = (lane >> 4) << 3;
#pragma unroll
    for (int j = 0; j < 4; ++j) {
        const int nSub = nBlk + wn * 64 + j * 16;
        if (nSub >= N) continue;
        float* crow = C + (size_t)(mBlk + wm * 16 + rb) * N + nSub + col;
#pragma unroll
        for (int r = 0; r < 8; ++r) crow[(size_t)r * N] = acc[j][r];
    }
}

// ---------------- elementwise f32 -> bf16 cast ----------------
__global__ void cast_bf16_kernel(const float* __restrict__ in, bf16* __restrict__ out, int n) {
    int i = blockIdx.x * blockDim.x + threadIdx.x;
    if (i < n) out[i] = f2bf(in[i]);
}

// ---------------- x_proj_w [33,2048] -> padded bf16 [64,2048] ----------------
__global__ void pad_xproj_kernel(const float* __restrict__ w, bf16* __restrict__ out) {
    int i = blockIdx.x * blockDim.x + threadIdx.x;   // over 64*2048
    if (i >= N_SSM_PAD * D_INNER) return;
    int e = i >> 11;
    int k = i & (D_INNER - 1);
    out[i] = (e < N_SSM) ? f2bf(w[e * D_INNER + k]) : f2bf(0.f);
}

// ---------------- depthwise causal conv + SiLU ----------------
// xz: [M_TOT, 4096]; x_in = cols [0,2048), z = cols [2048,4096)
__global__ __launch_bounds__(256) void conv_silu_kernel(
    const float* __restrict__ xz,
    const float* __restrict__ cw, const float* __restrict__ cb,
    float* __restrict__ xc_f32, bf16* __restrict__ xc_bf)
{
    int idx = blockIdx.x * blockDim.x + threadIdx.x;  // m*2048 + d
    if (idx >= M_TOT * D_INNER) return;
    int d = idx & (D_INNER - 1);
    int m = idx >> 11;
    int b = m >> 10;
    int l = m & (SEQ - 1);

    float s = cb[d];
#pragma unroll
    for (int j = 0; j < D_CONV; ++j) {
        int li = l - (D_CONV - 1) + j;
        if (li >= 0)
            s += xz[((size_t)(b * SEQ + li)) * N_XZ + d] * cw[d * D_CONV + j];
    }
    float v = s * (1.f / (1.f + __expf(-s)));   // silu
    xc_f32[idx] = v;
    xc_bf[idx]  = f2bf(v);
}

// ---------------- selective scan (recurrence form) + output gate ----------------
// ssm: [M_TOT, 64] with col0 = dt_raw, 1..16 = B, 17..32 = C
__global__ __launch_bounds__(256) void scan_kernel(
    const float* __restrict__ ssm,
    const float* __restrict__ xz,       // for z gate
    const float* __restrict__ xc,       // f32 x_c
    const float* __restrict__ dtw, const float* __restrict__ dtb,
    const float* __restrict__ A_log, const float* __restrict__ Dp,
    bf16* __restrict__ y_bf)
{
    int t = blockIdx.x * blockDim.x + threadIdx.x;   // over BATCH * D_INNER
    if (t >= BATCH * D_INNER) return;
    int b = t >> 11;
    int d = t & (D_INNER - 1);

    float a[D_STATE], h[D_STATE];
#pragma unroll
    for (int n = 0; n < D_STATE; ++n) {
        a[n] = -__expf(A_log[d * D_STATE + n]);
        h[n] = 0.f;
    }
    const float w = dtw[d], bias = dtb[d], Dd = Dp[d];

    for (int l = 0; l < SEQ; ++l) {
        size_t m = (size_t)b * SEQ + l;
        const float* srow = ssm + m * N_SSM_PAD;
        float xr = w * srow[0] + bias;
        float dt = (xr > 20.f) ? xr : __logf(1.f + __expf(xr));  // softplus
        float xcv = xc[m * D_INNER + d];
        float dtx = dt * xcv;
        float accv = 0.f;
#pragma unroll
        for (int n = 0; n < D_STATE; ++n) {
            h[n] = __expf(dt * a[n]) * h[n] + dtx * srow[1 + n];
            accv += srow[17 + n] * h[n];
        }
        float y  = accv + Dd * xcv;
        float zv = xz[m * N_XZ + D_INNER + d];
        y *= zv * (1.f / (1.f + __expf(-zv)));    // * silu(z)
        y_bf[m * D_INNER + d] = f2bf(y);
    }
}

extern "C" void kernel_launch(void* const* d_in, const int* in_sizes, int n_in,
                              void* d_out, int out_size, void* d_ws, size_t ws_size,
                              hipStream_t stream) {
    (void)in_sizes; (void)n_in; (void)out_size; (void)ws_size;

    const float* x        = (const float*)d_in[0];   // [2,1024,1024]
    const float* w_in     = (const float*)d_in[1];   // [4096,1024]
    const float* conv_w   = (const float*)d_in[2];   // [2048,1,4]
    const float* conv_b   = (const float*)d_in[3];   // [2048]
    const float* w_xproj  = (const float*)d_in[4];   // [33,2048]
    const float* dt_w     = (const float*)d_in[5];   // [2048,1]
    const float* dt_b     = (const float*)d_in[6];   // [2048]
    const float* A_log    = (const float*)d_in[7];   // [2048,16]
    const float* D_param  = (const float*)d_in[8];   // [2048]
    const float* w_out    = (const float*)d_in[9];   // [1024,2048]
    float* out            = (float*)d_out;           // [2,1024,1024]

    // ---- workspace carve-up ----
    char* ws = (char*)d_ws;
    size_t off = 0;
    auto carve = [&](size_t bytes) { char* p = ws + off; off += (bytes + 255) & ~(size_t)255; return p; };
    bf16*  x_bf    = (bf16*) carve((size_t)M_TOT * D_MODEL * 2);
    bf16*  w_in_bf = (bf16*) carve((size_t)N_XZ * D_MODEL * 2);
    bf16*  w_xp_bf = (bf16*) carve((size_t)N_SSM_PAD * D_INNER * 2);
    bf16*  w_out_bf= (bf16*) carve((size_t)D_MODEL * D_INNER * 2);
    float* xz      = (float*)carve((size_t)M_TOT * N_XZ * 4);
    float* xc_f32  = (float*)carve((size_t)M_TOT * D_INNER * 4);
    bf16*  xc_bf   = (bf16*) carve((size_t)M_TOT * D_INNER * 2);
    float* ssm     = (float*)carve((size_t)M_TOT * N_SSM_PAD * 4);
    bf16*  y_bf    = (bf16*) carve((size_t)M_TOT * D_INNER * 2);

    // ---- 1. casts / padding ----
    {
        int n = M_TOT * D_MODEL;
        cast_bf16_kernel<<<(n + 255) / 256, 256, 0, stream>>>(x, x_bf, n);
    }
    {
        int n = N_XZ * D_MODEL;
        cast_bf16_kernel<<<(n + 255) / 256, 256, 0, stream>>>(w_in, w_in_bf, n);
    }
    {
        int n = D_MODEL * D_INNER;
        cast_bf16_kernel<<<(n + 255) / 256, 256, 0, stream>>>(w_out, w_out_bf, n);
    }
    {
        int n = N_SSM_PAD * D_INNER;
        pad_xproj_kernel<<<(n + 255) / 256, 256, 0, stream>>>(w_xproj, w_xp_bf);
    }

    // ---- 2. xz = x @ in_proj_w^T : M=2048, N=4096, K=1024 ----
    gemm_bf16_wmma<<<dim3((N_XZ + N_BLK - 1) / N_BLK, M_TOT / M_BLK), 256, 0, stream>>>(
        x_bf, w_in_bf, xz, M_TOT, N_XZ, D_MODEL);

    // ---- 3. depthwise conv + SiLU ----
    {
        int n = M_TOT * D_INNER;
        conv_silu_kernel<<<(n + 255) / 256, 256, 0, stream>>>(
            xz, conv_w, conv_b, xc_f32, xc_bf);
    }

    // ---- 4. ssm_in = x_c @ x_proj_w^T (padded to N=64) : M=2048, N=64, K=2048 ----
    gemm_bf16_wmma<<<dim3((N_SSM_PAD + N_BLK - 1) / N_BLK, M_TOT / M_BLK), 256, 0, stream>>>(
        xc_bf, w_xp_bf, ssm, M_TOT, N_SSM_PAD, D_INNER);

    // ---- 5. selective scan + gate -> y (bf16) ----
    scan_kernel<<<(BATCH * D_INNER) / 256, 256, 0, stream>>>(
        ssm, xz, xc_f32, dt_w, dt_b, A_log, D_param, y_bf);

    // ---- 6. out = y @ out_proj_w^T : M=2048, N=1024, K=2048 ----
    gemm_bf16_wmma<<<dim3((D_MODEL + N_BLK - 1) / N_BLK, M_TOT / M_BLK), 256, 0, stream>>>(
        y_bf, w_out_bf, out, M_TOT, D_MODEL, D_INNER);
}
